// DGMBlock_40329742910073
// MI455X (gfx1250) — matrix operand order repaired and checked by reading, not compiled, not run on hardware
//
#include <hip/hip_runtime.h>

// ---------------- problem constants (match reference) ----------------
constexpr int Nn   = 12288;   // nodes
constexpr int DIN  = 256;     // input feature dim
constexpr int DOUT = 256;     // embedding channels
constexpr int EE   = 196608;  // edges
constexpr int KTOP = 4;       // kNN per node

// output regions (floats)
constexpr size_t E_OFF  = (size_t)Nn * DOUT;                // edges row 0: top idx flat
constexpr size_t E_OFF2 = E_OFF + (size_t)Nn * KTOP;        // edges row 1: repeated rows
constexpr size_t T_OFF  = E_OFF + (size_t)2 * Nn * KTOP;    // top_vals

// ---------------- types for WMMA ----------------
typedef __bf16 bf16t;
typedef bf16t v16bf __attribute__((ext_vector_type(16)));
typedef float v8f   __attribute__((ext_vector_type(8)));
typedef unsigned int u32x4 __attribute__((ext_vector_type(4)));
typedef int i32x4 __attribute__((ext_vector_type(4)));

union Frag { v16bf v; u32x4 u[2]; };

// Async global->LDS path (CDNA5 GLOBAL_LOAD_ASYNC_TO_LDS_B128, ASYNCcnt).
// Guarded so the file still compiles on toolchains without the builtins.
#if defined(__gfx1250__) && \
    __has_builtin(__builtin_amdgcn_global_load_async_to_lds_b128) && \
    __has_builtin(__builtin_amdgcn_s_wait_asynccnt)
#define HAVE_ASYNC_LDS 1
#endif

#define DEVFN static __device__ __forceinline__

DEVFN void copy16_g2l(const unsigned short* src, unsigned short* lds_dst) {
#ifdef HAVE_ASYNC_LDS
  // builtin signature (per hipcc diagnostic): (int4 AS1*, int4 AS3*, imm offset, cpol)
  __builtin_amdgcn_global_load_async_to_lds_b128(
      (__attribute__((address_space(1))) i32x4*)(const void*)src,
      (__attribute__((address_space(3))) i32x4*)(void*)lds_dst,
      /*imm offset*/0, /*cpol*/0);
#else
  *(u32x4*)lds_dst = *(const u32x4*)src;
#endif
}

DEVFN void async_fence_then_barrier() {
#ifdef HAVE_ASYNC_LDS
  __builtin_amdgcn_s_wait_asynccnt(0);
#endif
  __syncthreads();
}

DEVFN unsigned short f2bf(float f) {
  union { float f; unsigned u; } x; x.f = f;
  unsigned r = (x.u + 0x7FFFu + ((x.u >> 16) & 1u)) >> 16;  // RNE
  return (unsigned short)r;
}

DEVFN void top4_insert(float v, int j, float* tv, int* ti) {
  if (v > tv[3]) {
    tv[3] = v; ti[3] = j;
    if (tv[3] > tv[2]) { float a=tv[2]; tv[2]=tv[3]; tv[3]=a; int c=ti[2]; ti[2]=ti[3]; ti[3]=c; }
    if (tv[2] > tv[1]) { float a=tv[1]; tv[1]=tv[2]; tv[2]=a; int c=ti[1]; ti[1]=ti[2]; ti[2]=c; }
    if (tv[1] > tv[0]) { float a=tv[0]; tv[0]=tv[1]; tv[1]=a; int c=ti[0]; ti[0]=ti[1]; ti[1]=c; }
  }
}

// ---------------- kernel 1: zero accumulators, init degrees ----------------
__global__ void k_init(float* __restrict__ hacc, float* __restrict__ deg) {
  size_t i = (size_t)blockIdx.x * blockDim.x + threadIdx.x;
  if (i < (size_t)Nn * DOUT) hacc[i] = 0.0f;
  if (i < (size_t)Nn)        deg[i]  = 1.0f;   // self loop
}

// ---------------- kernel 2: degree histogram ----------------
__global__ void k_deg(const int* __restrict__ ei, float* __restrict__ deg) {
  int e = blockIdx.x * blockDim.x + threadIdx.x;
  if (e < EE) atomicAdd(&deg[ei[e]], 1.0f);   // ei[0:E] = destinations
}

// ---------------- kernel 3: dinv = rsqrt(max(deg,1)) ----------------
__global__ void k_dinv(const float* __restrict__ deg, float* __restrict__ dinv) {
  int i = blockIdx.x * blockDim.x + threadIdx.x;
  if (i < Nn) dinv[i] = rsqrtf(fmaxf(deg[i], 1.0f));
}

// ---------------- kernel 4: bf16 copies of x and W^T ----------------
__global__ void k_xbf(const float* __restrict__ x, unsigned short* __restrict__ xbf) {
  size_t i = (size_t)blockIdx.x * blockDim.x + threadIdx.x;
  if (i < (size_t)Nn * DIN) xbf[i] = f2bf(x[i]);
}

__global__ void k_wt(const float* __restrict__ W, unsigned short* __restrict__ Wt) {
  int idx = blockIdx.x * blockDim.x + threadIdx.x;
  if (idx < DIN * DOUT) {
    int n = idx >> 8, k = idx & 255;
    Wt[n * DIN + k] = f2bf(W[(size_t)k * DOUT + n]);
  }
}

// ---------------- kernel 5: hw = x @ W  (bf16 WMMA, f32 accum) ----------------
// block = 256 thr (8 waves) -> 32 rows x 64 cols; wave tile = 16x16, K loop 256/32
__global__ __launch_bounds__(256) void k_gemm_xw(const unsigned short* __restrict__ xbf,
                                                 const unsigned short* __restrict__ Wt,
                                                 float* __restrict__ hw) {
  constexpr int APITCH = 264;                    // ushorts per row (pad breaks bank pattern)
  __shared__ unsigned short sA[32 * APITCH];
  int row0 = blockIdx.x * 32, col0 = blockIdx.y * 64;
  // stage A tile: 32 rows x 256 bf16, as 1024 16B chunks (async to LDS if available)
  for (int t = threadIdx.x; t < 1024; t += 256) {
    int rr = t >> 5, cc = (t & 31) * 8;
    copy16_g2l(xbf + (size_t)(row0 + rr) * DIN + cc, sA + rr * APITCH + cc);
  }
  async_fence_then_barrier();

  int w = threadIdx.x >> 5, L = threadIdx.x & 31;
  int r = w >> 2, c = w & 3, hh = L >> 4, m16 = L & 15;
  v8f acc = {0.f,0.f,0.f,0.f,0.f,0.f,0.f,0.f};
  int col = col0 + c * 16 + m16;

  #pragma unroll
  for (int k0 = 0; k0 < DIN; k0 += 32) {
    Frag a, bb;
    const u32x4* pa = (const u32x4*)(sA + (r * 16 + m16) * APITCH + k0 + hh * 8);
    a.u[0] = pa[0];                 // K = k0 + hh*8 + 0..7
    a.u[1] = pa[2];                 // K = k0 + hh*8 + 16..23
    const u32x4* pb = (const u32x4*)(Wt + (size_t)col * DIN + k0 + hh * 8);
    bb.u[0] = pb[0];                // K = k0 + hh*8 + 0..7
    bb.u[1] = pb[2];                // K = k0 + hh*8 + 16..23
    acc = __builtin_amdgcn_wmma_f32_16x16x32_bf16(false, a.v, false, bb.v,
                                                  (short)0, acc, false, false);
  }
  #pragma unroll
  for (int v = 0; v < 8; v++) {
    int rowg = row0 + r * 16 + hh * 8 + v;
    hw[(size_t)rowg * DOUT + col] = acc[v];
  }
}

// ---------------- kernel 6: edge scatter with symmetric norm ----------------
__global__ __launch_bounds__(256) void k_scatter(const int* __restrict__ ei,
                                                 const float* __restrict__ hw,
                                                 const float* __restrict__ dinv,
                                                 float* __restrict__ hacc) {
  int e = blockIdx.x * 8 + (threadIdx.x >> 5);
  int lane = threadIdx.x & 31;
  if (e >= EE) return;
  int row = ei[e], col = ei[EE + e];
  float nrm = dinv[row] * dinv[col];
  const float* src = hw + (size_t)col * DOUT;
  float* dst = hacc + (size_t)row * DOUT;
  #pragma unroll
  for (int f = lane; f < DOUT; f += 32) atomicAdd(&dst[f], nrm * src[f]);
}

// ---------------- kernel 7: h = hacc + dinv^2*hw + b ; hbf ; sq ----------------
__global__ __launch_bounds__(256) void k_finalize(const float* __restrict__ hacc,
                                                  const float* __restrict__ hw,
                                                  const float* __restrict__ dinv,
                                                  const float* __restrict__ bias,
                                                  float* __restrict__ hOut,
                                                  unsigned short* __restrict__ hbf,
                                                  float* __restrict__ sq) {
  int row = blockIdx.x * 8 + (threadIdx.x >> 5);
  int lane = threadIdx.x & 31;
  if (row >= Nn) return;
  float dii = dinv[row]; dii *= dii;
  float s = 0.0f;
  #pragma unroll
  for (int f = lane; f < DOUT; f += 32) {
    size_t o = (size_t)row * DOUT + f;
    float v = hacc[o] + dii * hw[o] + bias[f];
    hOut[o] = v;
    hbf[o] = f2bf(v);
    s += v * v;
  }
  #pragma unroll
  for (int mask = 16; mask >= 1; mask >>= 1) s += __shfl_xor(s, mask, 32);
  if (lane == 0) sq[row] = s;
}

// ---------------- kernel 8: fused Gram (WMMA) + Gumbel + top-4 ----------------
// block = 256 thr: 32 rows; 8 waves each own a 16x16 tile (r = w/4, c = w%4);
// stream all 12288 columns in 64-wide slabs; only q is read at N*N scale.
__global__ __launch_bounds__(256) void k_gram_topk(const unsigned short* __restrict__ hbf,
                                                   const float* __restrict__ sq,
                                                   const float* __restrict__ q,
                                                   const float* __restrict__ temp,
                                                   float* __restrict__ out) {
  constexpr int APITCH = 264;
  __shared__ unsigned short sA[32 * APITCH];
  __shared__ float sSq[32];
  __shared__ float cval[32][4][KTOP];
  __shared__ int   cidx[32][4][KTOP];

  int row0 = blockIdx.x * 32;
  // stage A tile straight from bf16 h (async to LDS if available)
  for (int t = threadIdx.x; t < 1024; t += 256) {
    int rr = t >> 5, cc = (t & 31) * 8;
    copy16_g2l(hbf + (size_t)(row0 + rr) * DOUT + cc, sA + rr * APITCH + cc);
  }
  if (threadIdx.x < 32) sSq[threadIdx.x] = sq[row0 + threadIdx.x];
  async_fence_then_barrier();

  float T = temp[0];
  int w = threadIdx.x >> 5, L = threadIdx.x & 31;
  int r = w >> 2, c = w & 3, hh = L >> 4, m16 = L & 15;
  int rbase = row0 + r * 16 + hh * 8;        // +v gives my 8 global rows

  float tv[8][KTOP]; int ti[8][KTOP];
  #pragma unroll
  for (int v = 0; v < 8; v++)
    #pragma unroll
    for (int k = 0; k < KTOP; k++) { tv[v][k] = -__builtin_inff(); ti[v][k] = 0; }

  float sr[8];
  #pragma unroll
  for (int v = 0; v < 8; v++) sr[v] = sSq[r * 16 + hh * 8 + v];

  for (int j0 = 0; j0 < Nn; j0 += 64) {
    int col = j0 + c * 16 + m16;
    __builtin_prefetch(q + (size_t)rbase * Nn + col + 64, 0, 1);  // next slab of q

    v8f acc = {0.f,0.f,0.f,0.f,0.f,0.f,0.f,0.f};
    #pragma unroll
    for (int k0 = 0; k0 < DOUT; k0 += 32) {
      Frag a, bb;
      const u32x4* pa = (const u32x4*)(sA + (r * 16 + m16) * APITCH + k0 + hh * 8);
      a.u[0] = pa[0];
      a.u[1] = pa[2];
      const u32x4* pb = (const u32x4*)(hbf + (size_t)col * DOUT + k0 + hh * 8);
      bb.u[0] = pb[0];
      bb.u[1] = pb[2];
      acc = __builtin_amdgcn_wmma_f32_16x16x32_bf16(false, a.v, false, bb.v,
                                                    (short)0, acc, false, false);
    }
    float sc = sq[col];
    #pragma unroll
    for (int v = 0; v < 8; v++) {
      int rowg = rbase + v;
      float d2 = fmaxf(sr[v] + sc - 2.0f * acc[v], 0.0f);
      float qv = q[(size_t)rowg * Nn + col];
      float g  = __logf(-__logf(qv + 1e-8f));
      float lq = -T * d2 - g;
      top4_insert(lq, col, tv[v], ti[v]);
    }
  }

  // butterfly-merge top4 lists across the 16 column-lanes of each half
  #pragma unroll
  for (int mask = 1; mask < 16; mask <<= 1) {
    #pragma unroll
    for (int v = 0; v < 8; v++) {
      float ov[KTOP]; int oi[KTOP];
      #pragma unroll
      for (int k = 0; k < KTOP; k++) {
        ov[k] = __shfl_xor(tv[v][k], mask, 32);
        oi[k] = __shfl_xor(ti[v][k], mask, 32);
      }
      #pragma unroll
      for (int k = 0; k < KTOP; k++) top4_insert(ov[k], oi[k], tv[v], ti[v]);
    }
  }

  if (m16 == 0) {   // one lane per half holds the merged lists for its 8 rows
    #pragma unroll
    for (int v = 0; v < 8; v++) {
      int R = r * 16 + hh * 8 + v;
      #pragma unroll
      for (int k = 0; k < KTOP; k++) { cval[R][c][k] = tv[v][k]; cidx[R][c][k] = ti[v][k]; }
    }
  }
  __syncthreads();

  // final merge across the 4 column-waves; write edges + top_vals
  if (threadIdx.x < 32) {
    int R = threadIdx.x, rowg = row0 + R;
    float fv[KTOP]; int fi[KTOP];
    #pragma unroll
    for (int k = 0; k < KTOP; k++) { fv[k] = -__builtin_inff(); fi[k] = 0; }
    #pragma unroll
    for (int l = 0; l < 4; l++)
      #pragma unroll
      for (int k = 0; k < KTOP; k++) top4_insert(cval[R][l][k], cidx[R][l][k], fv, fi);
    #pragma unroll
    for (int k = 0; k < KTOP; k++) {
      out[E_OFF  + (size_t)rowg * KTOP + k] = (float)fi[k];   // edges row 0: indices
      out[E_OFF2 + (size_t)rowg * KTOP + k] = (float)rowg;    // edges row 1: source rows
      out[T_OFF  + (size_t)rowg * KTOP + k] = fv[k];          // top_vals
    }
  }
}

// ---------------- host launcher ----------------
extern "C" void kernel_launch(void* const* d_in, const int* in_sizes, int n_in,
                              void* d_out, int out_size, void* d_ws, size_t ws_size,
                              hipStream_t stream) {
  const float* x    = (const float*)d_in[0];
  const float* W    = (const float*)d_in[1];
  const float* bias = (const float*)d_in[2];
  const float* temp = (const float*)d_in[3];
  const float* q    = (const float*)d_in[4];
  const int*   ei   = (const int*)d_in[5];
  float* out = (float*)d_out;

  // workspace carve-up (~38 MB)
  float* hw   = (float*)d_ws;                          // [N, DOUT] xW
  float* hacc = hw   + (size_t)Nn * DOUT;              // [N, DOUT] scatter accum
  float* deg  = hacc + (size_t)Nn * DOUT;              // [N]
  float* dinv = deg  + Nn;                             // [N]
  float* sq   = dinv + Nn;                             // [N]
  unsigned short* hbf = (unsigned short*)(sq + Nn);    // [N, DOUT] bf16 h
  unsigned short* Wt  = hbf + (size_t)Nn * DOUT;       // [DOUT, DIN] bf16 W^T
  unsigned short* xbf = Wt  + (size_t)DIN * DOUT;      // [N, DIN]  bf16 x

  k_init    <<<(Nn * DOUT + 255) / 256, 256, 0, stream>>>(hacc, deg);
  k_deg     <<<(EE + 255) / 256,        256, 0, stream>>>(ei, deg);
  k_dinv    <<<(Nn + 255) / 256,        256, 0, stream>>>(deg, dinv);
  k_xbf     <<<(Nn * DIN + 255) / 256,  256, 0, stream>>>(x, xbf);
  k_wt      <<<(DIN * DOUT + 255) / 256,256, 0, stream>>>(W, Wt);
  k_gemm_xw <<<dim3(Nn / 32, DOUT / 64),256, 0, stream>>>(xbf, Wt, hw);
  k_scatter <<<EE / 8,                  256, 0, stream>>>(ei, hw, dinv, hacc);
  k_finalize<<<Nn / 8,                  256, 0, stream>>>(hacc, hw, dinv, bias, out, hbf, sq);
  k_gram_topk<<<Nn / 32,                256, 0, stream>>>(hbf, sq, q, temp, out);
}
